// Transformer_decoder_9646496546928
// MI455X (gfx1250) — matrix operand order repaired and checked by reading, not compiled, not run on hardware
//
#include <hip/hip_runtime.h>
#include <hip/hip_bf16.h>

typedef __attribute__((ext_vector_type(16))) _Float16 v16h;
typedef __attribute__((ext_vector_type(8)))  float    v8f;

namespace {

constexpr int kB = 32, kS = 512, kD = 512, kH = 8, kDK = 64, kL = 6, kDFF = 1024;
constexpr int kRows = kB * kS;   // 16384 token rows

struct GemmP {
  const _Float16* A;
  const _Float16* Bm;
  const float* bias;
  const float* resid;
  float* C32;
  _Float16* C16;
  _Float16* C16T;       // optional transposed f16 store: [(col-col0)*ldct + row]
  int M, N, K;
  int lda, ldb, ldc;
  int col0, ldct;
  float scale;
  int relu, causal, H;
  long long aB, aH, bB, bH, cB, cH;
};

__device__ inline float wave_sum(float v) {
#pragma unroll
  for (int o = 16; o > 0; o >>= 1) v += __shfl_xor(v, o, 32);
  return v;
}
__device__ inline float wave_max(float v) {
#pragma unroll
  for (int o = 16; o > 0; o >>= 1) v = fmaxf(v, __shfl_xor(v, o, 32));
  return v;
}

// One 16B segment per lane, global -> LDS, no VGPR round trip (ASYNCcnt).
__device__ inline void async_b128(_Float16* lds_dst, const _Float16* gsrc) {
  const unsigned int l = (unsigned int)(unsigned long long)lds_dst;
  asm volatile("global_load_async_to_lds_b128 %0, %1, off"
               :: "v"(l), "v"(gsrc)
               : "memory");
}

// ---------------------------------------------------------------------------
// Batched WMMA GEMM:  C = act( scale * (A @ B^T) + bias + resid )
//   A : f16, M x K, row stride lda;  Bm : (N,K) n-major rows, stride ldb.
//   128x64 block tile, 8 waves, each wave owns a 32x32 C block
//   -> 4 v_wmma_f32_16x16x32_f16 per k-step.
//   Double-buffered LDS tiles filled by global_load_async_to_lds_b128
//   (3 per wave per tile); s_wait_asynccnt 3 drains exactly the previous
//   tile while the next one is in flight.
// ---------------------------------------------------------------------------
__global__ __launch_bounds__(256) void wmma_gemm(GemmP p) {
  __shared__ _Float16 As[2][128][40];
  __shared__ _Float16 Bs[2][64][40];

  const int z  = blockIdx.z;
  const int zb = p.H ? z / p.H : 0;
  const int zh = p.H ? z % p.H : 0;
  const _Float16* __restrict__ A  = p.A  + zb * p.aB + zh * p.aH;
  const _Float16* __restrict__ Bm = p.Bm + zb * p.bB + zh * p.bH;
  const long long coff = zb * p.cB + zh * p.cH;

  const int m0 = blockIdx.y * 128, n0 = blockIdx.x * 64;
  const int t = threadIdx.x, lane = t & 31, w = t >> 5;
  const int wm = (w & 3) * 32;   // wave's 32-row strip of C
  const int wn = (w >> 2) * 32;  // wave's 32-col strip of C
  const int hf = lane >> 4, l16 = lane & 15;

  // staging geometry: A = 128x32 (two 16B segments/thread), B = 64x32 (one)
  const int ar0 = t >> 1;                       // segid t*2   -> rows 0..127
  const int as0 = (t & 1) << 4;                 // two 8-half segs per row pair
  // simpler split: segments t and t+256 of 512 total
  const int rowA0 = t >> 2, segA0 = (t & 3) << 3;          // rows 0..63
  const int rowA1 = 64 + (t >> 2), segA1 = (t & 3) << 3;   // rows 64..127
  const int rowB = t >> 2, segB = (t & 3) << 3;
  (void)ar0; (void)as0;
  // clamp OOB rows into the allocation; the epilogue bounds-check discards
  // them, and clamping keeps EXEC full so ASYNCcnt math stays exact.
  const int gmA0 = min(m0 + rowA0, p.M - 1);
  const int gmA1 = min(m0 + rowA1, p.M - 1);
  const int gnB  = min(n0 + rowB, p.N - 1);

  v8f acc00 = {}, acc01 = {}, acc10 = {}, acc11 = {};
  const int nk = p.K >> 5;

  auto issue_tile = [&](int kt, int buf) {
    const int k0 = kt << 5;
    async_b128(&As[buf][rowA0][segA0],
               A + (long long)gmA0 * p.lda + (k0 + segA0));
    async_b128(&As[buf][rowA1][segA1],
               A + (long long)gmA1 * p.lda + (k0 + segA1));
    async_b128(&Bs[buf][rowB][segB],
               Bm + (long long)gnB * p.ldb + (k0 + segB));
  };

  issue_tile(0, 0);
  for (int kt = 0; kt < nk; ++kt) {
    const int cb = kt & 1;
    if (kt + 1 < nk) {
      issue_tile(kt + 1, cb ^ 1);
      asm volatile("s_wait_asynccnt 3" ::: "memory");  // tile kt landed
    } else {
      asm volatile("s_wait_asynccnt 0" ::: "memory");
    }
    __syncthreads();  // all waves' tile kt visible in LDS

    // Fragments per ISA 7.12.2 wave32 layouts.
    v16h af0, af1, bf0, bf1;
#pragma unroll
    for (int j = 0; j < 8; ++j) {
      const int kb = ((j & 4) << 2) + (hf << 3) + ((j & 3) << 1);
      af0[2 * j]     = As[cb][wm + l16][kb];
      af0[2 * j + 1] = As[cb][wm + l16][kb + 1];
      af1[2 * j]     = As[cb][wm + 16 + l16][kb];
      af1[2 * j + 1] = As[cb][wm + 16 + l16][kb + 1];
    }
#pragma unroll
    for (int v = 0; v < 8; ++v) {
      const int kk2 = ((v & 4) << 2) + (hf << 3) + ((v & 3) << 1);
      bf0[2 * v]     = Bs[cb][wn + l16][kk2];
      bf0[2 * v + 1] = Bs[cb][wn + l16][kk2 + 1];
      bf1[2 * v]     = Bs[cb][wn + 16 + l16][kk2];
      bf1[2 * v + 1] = Bs[cb][wn + 16 + l16][kk2 + 1];
    }
    acc00 = __builtin_amdgcn_wmma_f32_16x16x32_f16(false, af0, false, bf0,
                                                   (short)0, acc00, false, false);
    acc01 = __builtin_amdgcn_wmma_f32_16x16x32_f16(false, af0, false, bf1,
                                                   (short)0, acc01, false, false);
    acc10 = __builtin_amdgcn_wmma_f32_16x16x32_f16(false, af1, false, bf0,
                                                   (short)0, acc10, false, false);
    acc11 = __builtin_amdgcn_wmma_f32_16x16x32_f16(false, af1, false, bf1,
                                                   (short)0, acc11, false, false);
    __syncthreads();  // tile cb consumed; safe to overwrite next iteration
  }

  // Epilogue: C VGPR r of subtile (sm,u) -> row wm+16*sm+8*hf+r, col +16*u+l16
#pragma unroll
  for (int sm = 0; sm < 2; ++sm) {
#pragma unroll
    for (int r = 0; r < 8; ++r) {
      const int grow = m0 + wm + sm * 16 + hf * 8 + r;
      if (grow >= p.M) continue;
#pragma unroll
      for (int u = 0; u < 2; ++u) {
        const int gcol = n0 + wn + u * 16 + l16;
        if (gcol >= p.N) continue;
        const v8f& a = sm ? (u ? acc11 : acc10) : (u ? acc01 : acc00);
        float val = a[r] * p.scale;
        if (p.bias) val += p.bias[gcol];
        if (p.causal && gcol > grow) val = -1.0e9f;  // reference tril mask
        const long long idx = coff + (long long)grow * p.ldc + gcol;
        if (p.resid) val += p.resid[idx];
        if (p.relu)  val = fmaxf(val, 0.0f);
        if (p.C32) p.C32[idx] = val;
        if (p.C16) p.C16[idx] = (_Float16)val;
        if (p.C16T && gcol >= p.col0)
          p.C16T[(long long)(gcol - p.col0) * p.ldct + grow] = (_Float16)val;
      }
    }
  }
}

// ---- LayerNorm: one wave per 512-wide row, writes f16 activations ---------
__global__ __launch_bounds__(256) void ln_k(const float* __restrict__ X,
                                            const float* __restrict__ g,
                                            const float* __restrict__ b,
                                            _Float16* __restrict__ O, int rows) {
  const int row = blockIdx.x * (blockDim.x >> 5) + (threadIdx.x >> 5);
  if (row >= rows) return;
  const int lane = threadIdx.x & 31;
  const float* x = X + (long long)row * kD;
  float v[16]; float s = 0.f;
#pragma unroll
  for (int i = 0; i < 16; ++i) { v[i] = x[lane + 32 * i]; s += v[i]; }
  s = wave_sum(s);
  const float mean = s * (1.0f / kD);
  float q = 0.f;
#pragma unroll
  for (int i = 0; i < 16; ++i) { float d = v[i] - mean; q += d * d; }
  q = wave_sum(q);
  const float inv = rsqrtf(q * (1.0f / kD) + 1e-5f);
  _Float16* o = O + (long long)row * kD;
#pragma unroll
  for (int i = 0; i < 16; ++i) {
    const int c = lane + 32 * i;
    o[c] = (_Float16)((v[i] - mean) * inv * g[c] + b[c]);
  }
}

// ---- Softmax over 512-wide score rows; writes f32 (attn out) + f16 (P) ----
__global__ __launch_bounds__(256) void softmax_k(float* __restrict__ P,
                                                 _Float16* __restrict__ P16,
                                                 long long rows) {
  const long long row =
      (long long)blockIdx.x * (blockDim.x >> 5) + (threadIdx.x >> 5);
  if (row >= rows) return;
  const int lane = threadIdx.x & 31;
  float* x = P + row * kS;
  float v[16]; float mx = -3.4e38f;
#pragma unroll
  for (int i = 0; i < 16; ++i) { v[i] = x[lane + 32 * i]; mx = fmaxf(mx, v[i]); }
  mx = wave_max(mx);
  float s = 0.f;
#pragma unroll
  for (int i = 0; i < 16; ++i) { v[i] = __expf(v[i] - mx); s += v[i]; }
  s = wave_sum(s);
  const float inv = 1.0f / s;
  _Float16* o = P16 + row * kS;
#pragma unroll
  for (int i = 0; i < 16; ++i) {
    const int c = lane + 32 * i;
    const float pv = v[i] * inv;
    x[c] = pv;
    o[c] = (_Float16)pv;
  }
}

// f32 -> f16, 4 elements per thread (16B load / 8B store)
__global__ void f32_to_f16_k(const float4* __restrict__ in,
                             ushort4* __restrict__ out, long long n4) {
  long long i = (long long)blockIdx.x * blockDim.x + threadIdx.x;
  const long long stride = (long long)gridDim.x * blockDim.x;
  for (; i < n4; i += stride) {
    const float4 f = in[i];
    _Float16 h[4] = {(_Float16)f.x, (_Float16)f.y, (_Float16)f.z,
                     (_Float16)f.w};
    out[i] = *(const ushort4*)h;
  }
}

// Gather arg embeddings into the (16384 x 1024) f16 A matrix, 4-wide.
__global__ void embed_gather_k(const int* __restrict__ x,
                               const float* __restrict__ argEmb,
                               _Float16* __restrict__ A16) {
  const long long i4 = (long long)blockIdx.x * blockDim.x + threadIdx.x;
  if (i4 >= (long long)kRows * 256) return;
  const int jq = (int)(i4 & 255);              // 4-elem group within 1024
  const long long r = i4 >> 8;                 // r = b*S + s
  const int b = (int)(r >> 9), s = (int)(r & 511);
  const int a = jq >> 5, c4 = (jq & 31) << 2;
  const int arg = x[((long long)s * kB + b) * 9 + 1 + a];
  const float4 f = *(const float4*)&argEmb[arg * 128 + c4];
  _Float16 h[4] = {(_Float16)f.x, (_Float16)f.y, (_Float16)f.z, (_Float16)f.w};
  *(ushort4*)&A16[i4 << 2] = *(const ushort4*)h;
}

// h += cmd_emb[cmd] + pos_encoding(s); row s==0 overwritten with cls_emb.
__global__ void embellish_k(float* __restrict__ h, const int* __restrict__ x,
                            const int* __restrict__ trg,
                            const float* __restrict__ cmdEmb,
                            const float* __restrict__ clsEmb) {
  const long long i = (long long)blockIdx.x * blockDim.x + threadIdx.x;
  if (i >= (long long)kRows * kD) return;
  const int d = (int)(i & 511);
  const long long r = i >> 9;                  // r = b*S + s
  const int b = (int)(r >> 9), s = (int)(r & 511);
  if (s == 0) { h[i] = clsEmb[(long long)trg[b] * kD + d]; return; }
  const int cmd = x[((long long)s * kB + b) * 9];
  const float dv = __expf((float)(d & ~1) * (-9.210340371976184f / (float)kD));
  const float ang = (float)s * dv;
  const float pe = (d & 1) ? __cosf(ang) : __sinf(ang);
  h[i] += cmdEmb[cmd * kD + d] + pe;
}

// h[b,s,:] += cv[b,:]  (collapsed cross-attention: Sk==1 -> softmax==1)
__global__ void addcv_k(float* __restrict__ h, const float* __restrict__ cv) {
  const long long i = (long long)blockIdx.x * blockDim.x + threadIdx.x;
  if (i >= (long long)kRows * kD) return;
  const int d = (int)(i & 511);
  const int b = (int)((i >> 9) >> 9);
  h[i] += cv[b * kD + d];
}

}  // namespace

extern "C" void kernel_launch(void* const* d_in, const int* in_sizes, int n_in,
                              void* d_out, int out_size, void* d_ws,
                              size_t ws_size, hipStream_t stream) {
  (void)in_sizes; (void)n_in; (void)out_size; (void)ws_size;
  const int*   x       = (const int*)  d_in[0];
  const float* memory  = (const float*)d_in[1];
  const int*   trg     = (const int*)  d_in[2];
  /* d_in[3] tgt_mask is tril(ones) -> folded into causal epilogue */
  const float* cmd_emb = (const float*)d_in[4];
  const float* arg_emb = (const float*)d_in[5];
  const float* efw     = (const float*)d_in[6];
  const float* efb     = (const float*)d_in[7];
  const float* sa_w    = (const float*)d_in[8];
  const float* sa_b    = (const float*)d_in[9];
  const float* ca_w    = (const float*)d_in[10];
  const float* ca_b    = (const float*)d_in[11];
  const float* ff_w1   = (const float*)d_in[12];
  const float* ff_b1   = (const float*)d_in[13];
  const float* ff_w2   = (const float*)d_in[14];
  const float* ff_b2   = (const float*)d_in[15];
  const float* ln_g    = (const float*)d_in[16];
  const float* ln_b    = (const float*)d_in[17];
  const float* fn_g    = (const float*)d_in[18];
  const float* fn_b    = (const float*)d_in[19];
  const float* cls_emb = (const float*)d_in[20];
  const float* cmd_w   = (const float*)d_in[21];
  const float* cmd_b   = (const float*)d_in[22];
  const float* argf_w  = (const float*)d_in[23];
  const float* argf_b  = (const float*)d_in[24];

  float* cmd_logits  = (float*)d_out;                          // (B*S, 4)
  float* args_logits = cmd_logits + (long long)kRows * 4;      // (B*S, 1024)
  float* attnP       = args_logits + (long long)kRows * 1024;  // (B,H,S,S)

  char* cur = (char*)d_ws;
  auto take = [&](long long bytes) -> char* {
    char* r = cur; cur += (bytes + 255) & ~255LL; return r;
  };
  float*    h       = (float*)   take((long long)kRows * kD * 4);
  _Float16* xn16    = (_Float16*)take((long long)kRows * kD * 2);
  _Float16* qkv16   = (_Float16*)take((long long)kRows * 3 * kD * 2);
  _Float16* vT16    = (_Float16*)take((long long)kD * kRows * 2);  // V^T
  _Float16* p16     = (_Float16*)take((long long)kB * kH * kS * kS * 2);
  _Float16* attno16 = (_Float16*)take((long long)kRows * kD * 2);
  _Float16* saw16   = (_Float16*)take((long long)kL * 4 * kD * kD * 2);
  _Float16* caw16   = (_Float16*)take((long long)kL * 4 * kD * kD * 2);
  _Float16* f1w16   = (_Float16*)take((long long)kL * kDFF * kD * 2);
  _Float16* f2w16   = (_Float16*)take((long long)kL * kD * kDFF * 2);
  _Float16* efw16   = (_Float16*)take((long long)kD * 1024 * 2);
  _Float16* agw16   = (_Float16*)take((long long)1024 * kD * 2);
  _Float16* cmw16   = (_Float16*)take(4 * kD * 2);
  _Float16* mem16   = (_Float16*)take(kB * kD * 2);
  _Float16* t116    = (_Float16*)take(kB * kD * 2);
  float*    cv      = (float*)   take((long long)kL * kB * kD * 4);
  _Float16* a16emb  = qkv16;  // alias: used before any QKV work
  _Float16* relu16  = qkv16;  // alias: used after attention is done

  auto conv = [&](const float* src, _Float16* dst, long long n) {
    const long long n4 = n >> 2;
    long long blocks = (n4 + 255) / 256; if (blocks > 65535) blocks = 65535;
    f32_to_f16_k<<<dim3((unsigned)blocks), 256, 0, stream>>>(
        (const float4*)src, (ushort4*)dst, n4);
  };
  auto gemm = [&](const _Float16* A, const _Float16* Bm, const float* bias,
                  const float* resid, float* C32, _Float16* C16,
                  _Float16* C16T, int M, int N, int K, int lda, int ldb,
                  int ldc, int col0, int ldct, float scale, int relu,
                  int causal, int batch, int Hh, long long aB, long long aH,
                  long long bB, long long bH, long long cB, long long cH) {
    GemmP p{A, Bm, bias, resid, C32, C16, C16T, M, N, K, lda, ldb, ldc,
            col0, ldct, scale, relu, causal, Hh, aB, aH, bB, bH, cB, cH};
    dim3 g((unsigned)((N + 63) / 64), (unsigned)((M + 127) / 128),
           (unsigned)batch);
    wmma_gemm<<<g, 256, 0, stream>>>(p);
  };

  // ---- per-call f16 weight copies ------------------------------------------
  conv(sa_w,   saw16, (long long)kL * 4 * kD * kD);
  conv(ca_w,   caw16, (long long)kL * 4 * kD * kD);
  conv(ff_w1,  f1w16, (long long)kL * kDFF * kD);
  conv(ff_w2,  f2w16, (long long)kL * kD * kDFF);
  conv(efw,    efw16, (long long)kD * 1024);
  conv(argf_w, agw16, (long long)1024 * kD);
  conv(cmd_w,  cmw16, 4 * kD);
  conv(memory, mem16, kB * kD);

  // ---- embedding: h = argEmbGather @ efw^T + efb, + cmd_emb + pos, cls row --
  embed_gather_k<<<(kRows * 256) / 256, 256, 0, stream>>>(x, arg_emb, a16emb);
  gemm(a16emb, efw16, efb, nullptr, h, nullptr, nullptr, kRows, kD, 1024,
       1024, 1024, kD, 0, 0, 1.f, 0, 0, 1, 0, 0, 0, 0, 0, 0, 0);
  embellish_k<<<(kRows * kD) / 256, 256, 0, stream>>>(h, x, trg, cmd_emb,
                                                      cls_emb);

  // ---- cross-attention collapses (Sk==1): cv[l] = (mem@Wv^T+bv)@Wo^T+bo ----
  for (int l = 0; l < kL; ++l) {
    gemm(mem16, caw16 + ((long long)l * 4 + 2) * kD * kD,
         ca_b + ((long long)l * 4 + 2) * kD, nullptr, nullptr, t116, nullptr,
         kB, kD, kD, kD, kD, kD, 0, 0, 1.f, 0, 0, 1, 0, 0, 0, 0, 0, 0, 0);
    gemm(t116, caw16 + ((long long)l * 4 + 3) * kD * kD,
         ca_b + ((long long)l * 4 + 3) * kD, nullptr,
         cv + (long long)l * kB * kD, nullptr, nullptr, kB, kD, kD, kD, kD,
         kD, 0, 0, 1.f, 0, 0, 1, 0, 0, 0, 0, 0, 0, 0);
  }

  // ---- decoder layers -------------------------------------------------------
  for (int l = 0; l < kL; ++l) {
    ln_k<<<kRows / 8, 256, 0, stream>>>(h, ln_g + (long long)(l * 3) * kD,
                                        ln_b + (long long)(l * 3) * kD, xn16,
                                        kRows);
    // fused QKV projection; V columns [1024,1536) also stored transposed
    gemm(xn16, saw16 + (long long)l * 4 * kD * kD,
         sa_b + (long long)l * 4 * kD, nullptr, nullptr, qkv16, vT16, kRows,
         3 * kD, kD, kD, kD, 3 * kD, 1024, kRows, 1.f, 0, 0, 1, 0, 0, 0, 0, 0,
         0, 0);
    // scores = Q @ K^T / sqrt(DK), causal; written into d_out attn region
    gemm(qkv16, qkv16 + kD, nullptr, nullptr, attnP, nullptr, nullptr, kS, kS,
         kDK, 3 * kD, 3 * kD, kS, 0, 0, 0.125f, 0, 1, kB * kH, kH,
         (long long)kS * 3 * kD, 64, (long long)kS * 3 * kD, 64,
         (long long)kH * kS * kS, (long long)kS * kS);
    // softmax rows (B*H*S of width S): f32 probs in place + f16 copy
    softmax_k<<<(kB * kH * kS) / 8, 256, 0, stream>>>(
        attnP, p16, (long long)kB * kH * kS);
    // O = P @ V using V^T (n-major rows d of length S, ldb = 16384)
    gemm(p16, vT16, nullptr, nullptr, nullptr, attno16, nullptr, kS, kDK, kS,
         kS, kRows, kD, 0, 0, 1.f, 0, 0, kB * kH, kH,
         (long long)kH * kS * kS, (long long)kS * kS,
         512, (long long)64 * kRows,
         (long long)kS * kD, 64);
    // out-proj + residual into h
    gemm(attno16, saw16 + ((long long)l * 4 + 3) * kD * kD,
         sa_b + ((long long)l * 4 + 3) * kD, h, h, nullptr, nullptr, kRows,
         kD, kD, kD, kD, kD, 0, 0, 1.f, 0, 0, 1, 0, 0, 0, 0, 0, 0, 0);
    addcv_k<<<(kRows * kD) / 256, 256, 0, stream>>>(h,
                                                    cv + (long long)l * kB * kD);
    ln_k<<<kRows / 8, 256, 0, stream>>>(h, ln_g + (long long)(l * 3 + 2) * kD,
                                        ln_b + (long long)(l * 3 + 2) * kD,
                                        xn16, kRows);
    gemm(xn16, f1w16 + (long long)l * kDFF * kD, ff_b1 + (long long)l * kDFF,
         nullptr, nullptr, relu16, nullptr, kRows, kDFF, kD, kD, kD, kDFF, 0,
         0, 1.f, 1, 0, 1, 0, 0, 0, 0, 0, 0, 0);
    gemm(relu16, f2w16 + (long long)l * kD * kDFF, ff_b2 + (long long)l * kD,
         h, h, nullptr, nullptr, kRows, kD, kDFF, kDFF, kDFF, kD, 0, 0, 1.f,
         0, 0, 1, 0, 0, 0, 0, 0, 0, 0);
  }

  // ---- final LN + heads -----------------------------------------------------
  ln_k<<<kRows / 8, 256, 0, stream>>>(h, fn_g, fn_b, xn16, kRows);
  gemm(xn16, cmw16, cmd_b, nullptr, cmd_logits, nullptr, nullptr, kRows, 4,
       kD, kD, kD, 4, 0, 0, 1.f, 0, 0, 1, 0, 0, 0, 0, 0, 0, 0);
  gemm(xn16, agw16, argf_b, nullptr, args_logits, nullptr, nullptr, kRows,
       1024, kD, kD, kD, 1024, 0, 0, 1.f, 0, 0, 1, 0, 0, 0, 0, 0, 0, 0);
}
